// TransformerBlockPipe_29566554866236
// MI455X (gfx1250) — matrix operand order repaired and checked by reading, not compiled, not run on hardware
//
#include <hip/hip_runtime.h>

typedef __bf16 bf16;
typedef __attribute__((ext_vector_type(16))) __bf16 v16bf;
typedef __attribute__((ext_vector_type(8)))  __bf16 v8bf;
typedef __attribute__((ext_vector_type(8)))  float  v8f;

#define DEV __device__ __forceinline__

static constexpr int kB  = 2;
static constexpr int kT  = 2048;
static constexpr int kD  = 1024;
static constexpr int kH  = 16;
static constexpr int kHd = 64;
static constexpr int kF  = 4096;              // 4*D
static constexpr int kTok = kB * kT;          // 4096 rows total

// ---------------------------------------------------------------------------
// WMMA helpers (CDNA5 wave32 layouts, 16x16x32 bf16 -> f32)
// ---------------------------------------------------------------------------
DEV v8f zero_v8f() {
  v8f z;
#pragma unroll
  for (int i = 0; i < 8; ++i) z[i] = 0.0f;
  return z;
}

DEV v16bf load_a_frag(const bf16* s, int row, int ld, int kbase) {
  const int lane = threadIdx.x & 31;
  const int m = lane & 15, half = lane >> 4;
  const bf16* p0 = s + (size_t)(row + m) * ld + kbase + half * 8;
  const bf16* p1 = p0 + 16;
  v16bf a;
#pragma unroll
  for (int j = 0; j < 8; ++j) { a[j] = p0[j]; a[8 + j] = p1[j]; }
  return a;
}

DEV v16bf load_b_frag(const bf16* s, int nbase, int ld, int kbase) {
  const int lane = threadIdx.x & 31;
  const int n = lane & 15, half = lane >> 4;
  const bf16* p0 = s + (size_t)(nbase + n) * ld + kbase + half * 8;
  const bf16* p1 = p0 + 16;
  v16bf b;
#pragma unroll
  for (int j = 0; j < 8; ++j) { b[j] = p0[j]; b[8 + j] = p1[j]; }
  return b;
}

DEV v8f wmma_bf16(v16bf a, v16bf b, v8f c) {
  return __builtin_amdgcn_wmma_f32_16x16x32_bf16(
      false, a, false, b, (short)0, c, false, false);
}

// Async 16-byte copy global -> LDS (CDNA5 GLOBAL_LOAD_ASYNC_TO_LDS_B128),
// tracked by ASYNCcnt; no VGPR round trip.
DEV void async_cp16(bf16* lds_dst, const bf16* gsrc) {
  unsigned lds = (unsigned)(size_t)lds_dst;                 // LDS aperture: low 32 bits
  unsigned long long ga = (unsigned long long)(size_t)gsrc;
  asm volatile("global_load_async_to_lds_b128 %0, %1, off"
               :: "v"(lds), "v"(ga) : "memory");
}
DEV void async_wait() {
  asm volatile("s_wait_asynccnt 0x0" ::: "memory");
}

// ---------------------------------------------------------------------------
// Batched tiled transpose + fp32->bf16: out[b][n][k] = in[b][k][n]
// grid = (K/32, N/32, batch), block = 256
// ---------------------------------------------------------------------------
__global__ void __launch_bounds__(256) k_transpose_bf16(const float* __restrict__ in,
                                                        bf16* __restrict__ out,
                                                        int K, int N) {
  __shared__ float tile[32][33];
  const float* ib = in + (size_t)blockIdx.z * K * N;
  bf16* ob = out + (size_t)blockIdx.z * K * N;
  const int k0 = blockIdx.x * 32, n0 = blockIdx.y * 32;
  const int tx = threadIdx.x & 31, ty = threadIdx.x >> 5;  // 32 x 8
#pragma unroll
  for (int i = ty; i < 32; i += 8)
    tile[i][tx] = ib[(size_t)(k0 + i) * N + (n0 + tx)];
  __syncthreads();
#pragma unroll
  for (int i = ty; i < 32; i += 8)
    ob[(size_t)(n0 + i) * K + (k0 + tx)] = (bf16)tile[tx][i];
}

// ---------------------------------------------------------------------------
// Weightless RMSNorm: one row (D=1024) per 256-thread block, output bf16
// ---------------------------------------------------------------------------
__global__ void __launch_bounds__(256) k_rmsnorm_bf16(const float* __restrict__ x,
                                                      bf16* __restrict__ o) {
  const int row = blockIdx.x;
  const int tid = threadIdx.x;
  const float* xr = x + (size_t)row * kD;
  float v[4];
  float ss = 0.f;
#pragma unroll
  for (int i = 0; i < 4; ++i) {
    v[i] = xr[tid + i * 256];
    ss += v[i] * v[i];
  }
#pragma unroll
  for (int m = 16; m >= 1; m >>= 1) ss += __shfl_xor(ss, m, 32);
  __shared__ float wsum[8];
  if ((tid & 31) == 0) wsum[tid >> 5] = ss;
  __syncthreads();
  float tot = 0.f;
#pragma unroll
  for (int i = 0; i < 8; ++i) tot += wsum[i];
  const float inv = rsqrtf(tot * (1.0f / kD) + 1e-6f);
  bf16* orow = o + (size_t)row * kD;
#pragma unroll
  for (int i = 0; i < 4; ++i) orow[tid + i * 256] = (bf16)(v[i] * inv);
}

// ---------------------------------------------------------------------------
// QKV projection: per (b,h,sel) GEMM  [T,1024] x [1024,64] -> bf16
// Double-buffered async-LDS staging. V output stored transposed [d][t].
// grid = (T/128, 3, B*H), block = 256
// ---------------------------------------------------------------------------
__global__ void __launch_bounds__(256) k_qkv(const bf16* __restrict__ xn,
                                             const bf16* __restrict__ wqt,
                                             const bf16* __restrict__ wkt,
                                             const bf16* __restrict__ wvt,
                                             bf16* __restrict__ qb,
                                             bf16* __restrict__ kb,
                                             bf16* __restrict__ vtb) {
  __shared__ bf16 As[2][128][32];
  __shared__ bf16 Bs[2][64][32];
  const int tid = threadIdx.x, w = tid >> 5, lane = tid & 31;
  const int n16 = lane & 15, hc = lane >> 4;
  const int z = blockIdx.z, b = z >> 4, hh = z & 15;
  const int sel = blockIdx.y;
  const bf16* A = xn + (size_t)b * kT * kD;
  const bf16* Wt = (sel == 0 ? wqt : sel == 1 ? wkt : wvt) + (size_t)hh * kD * kHd;
  const int m0 = blockIdx.x * 128;

  auto stage = [&](int buf, int k0) {
#pragma unroll
    for (int l = 0; l < 2; ++l) {                // A: 128x32 in 16B chunks
      int ch = tid + l * 256;
      int r = ch >> 2, c8 = (ch & 3) * 8;
      async_cp16(&As[buf][r][c8], &A[(size_t)(m0 + r) * kD + k0 + c8]);
    }
    {                                            // B: 64x32 ([n][k])
      int r = tid >> 2, c8 = (tid & 3) * 8;
      async_cp16(&Bs[buf][r][c8], &Wt[(size_t)r * kD + k0 + c8]);
    }
  };

  v8f acc[4];
#pragma unroll
  for (int i = 0; i < 4; ++i) acc[i] = zero_v8f();

  stage(0, 0);
  async_wait();
  __syncthreads();
  constexpr int nIt = kD / 32;
  for (int it = 0; it < nIt; ++it) {
    const int cur = it & 1;
    if (it + 1 < nIt) stage(1 - cur, (it + 1) * 32);   // prefetch next tile async
    v16bf a = load_a_frag(&As[cur][0][0], w * 16, 32, 0);
    v16bf bk[4];
#pragma unroll
    for (int nt = 0; nt < 4; ++nt) bk[nt] = load_b_frag(&Bs[cur][0][0], nt * 16, 32, 0);
#pragma unroll
    for (int nt = 0; nt < 4; ++nt) acc[nt] = wmma_bf16(a, bk[nt], acc[nt]);
    async_wait();
    __syncthreads();
  }

  if (sel == 2) {                                // V stored transposed [d][t]
    bf16* O = vtb + (size_t)z * kT * kHd;
#pragma unroll
    for (int nt = 0; nt < 4; ++nt)
#pragma unroll
      for (int e = 0; e < 8; ++e) {
        int row = m0 + w * 16 + e + 8 * hc;
        int col = nt * 16 + n16;
        O[(size_t)col * kT + row] = (bf16)acc[nt][e];
      }
  } else {
    bf16* O = (sel == 0 ? qb : kb) + (size_t)z * kT * kHd;
#pragma unroll
    for (int nt = 0; nt < 4; ++nt)
#pragma unroll
      for (int e = 0; e < 8; ++e) {
        int row = m0 + w * 16 + e + 8 * hc;
        int col = nt * 16 + n16;
        O[(size_t)row * kHd + col] = (bf16)acc[nt][e];
      }
  }
}

// ---------------------------------------------------------------------------
// Causal flash attention per (b,h), async-LDS staged K/V tiles.
// grid = (T/128, 1, B*H), block = 256 (8 waves x 16 query rows).
// Writes h = x_ori + attn_out.
// ---------------------------------------------------------------------------
__global__ void __launch_bounds__(256) k_attn(const bf16* __restrict__ q,
                                              const bf16* __restrict__ kmat,
                                              const bf16* __restrict__ vtmat,
                                              const float* __restrict__ x_ori,
                                              float* __restrict__ h) {
  __shared__ bf16 Ks[64][64];         // [key][d]  (B frag: n=key, k=d)
  __shared__ bf16 Vs[64][64];         // [d][key]  (B frag: n=d,  k=key)
  __shared__ float Ssc[8][16][64];    // per-wave score scratch (aliased as bf16 P)
  __shared__ float corrw[8][16];
  __shared__ float lw[8][16];

  const int tid = threadIdx.x, w = tid >> 5, lane = tid & 31;
  const int n16 = lane & 15, hc = lane >> 4;
  const int z = blockIdx.z, b = z >> 4, hh = z & 15;
  const int q0 = blockIdx.x * 128 + w * 16;
  const bf16* qg = q + (size_t)z * kT * kHd;
  const bf16* kg = kmat + (size_t)z * kT * kHd;
  const bf16* vg = vtmat + (size_t)z * kT * kHd;   // [d][t]

  v16bf aq0 = load_a_frag(qg, q0, kHd, 0);
  v16bf aq1 = load_a_frag(qg, q0, kHd, 32);

  v8f accO[4];
#pragma unroll
  for (int i = 0; i < 4; ++i) accO[i] = zero_v8f();
  float m_run = -1e30f, l_run = 0.f;
  const int r = n16;
  const int nkt = blockIdx.x * 2 + 2;

  for (int kt = 0; kt < nkt; ++kt) {
    __syncthreads();
    // ---- async stage K (row-major) and V (pre-transposed) tiles ----
#pragma unroll
    for (int l = 0; l < 2; ++l) {
      int ch = tid + l * 256;
      int key = ch >> 3, c8 = (ch & 7) * 8;
      async_cp16(&Ks[key][c8], &kg[(size_t)(kt * 64 + key) * kHd + c8]);
    }
#pragma unroll
    for (int l = 0; l < 2; ++l) {
      int ch = tid + l * 256;
      int d = ch >> 3, c8 = (ch & 7) * 8;
      async_cp16(&Vs[d][c8], &vg[(size_t)d * kT + kt * 64 + c8]);
    }
    if (kt + 1 < nkt) {       // near-cache prefetch of next tile
      int rr = tid >> 2;
      __builtin_prefetch(&kg[(size_t)((kt + 1) * 64 + rr) * kHd], 0, 3);
      __builtin_prefetch(&vg[(size_t)rr * kT + (kt + 1) * 64], 0, 3);
    }
    async_wait();
    __syncthreads();
    if (kt * 64 > q0 + 15) continue;

    // ---- scores S = (Q K^T) / 8 ----
    v8f accS[4];
#pragma unroll
    for (int i = 0; i < 4; ++i) accS[i] = zero_v8f();
    {
      v16bf bk[8];
#pragma unroll
      for (int nt = 0; nt < 4; ++nt) {
        bk[2 * nt]     = load_b_frag(&Ks[0][0], nt * 16, 64, 0);
        bk[2 * nt + 1] = load_b_frag(&Ks[0][0], nt * 16, 64, 32);
      }
#pragma unroll
      for (int nt = 0; nt < 4; ++nt) {
        accS[nt] = wmma_bf16(aq0, bk[2 * nt], accS[nt]);
        accS[nt] = wmma_bf16(aq1, bk[2 * nt + 1], accS[nt]);
      }
    }
#pragma unroll
    for (int nt = 0; nt < 4; ++nt)
#pragma unroll
      for (int e = 0; e < 8; ++e)
        Ssc[w][e + 8 * hc][nt * 16 + n16] = accS[nt][e] * 0.125f;

    // ---- online softmax: lane -> row r, cols hc*32..+31 ----
    float sv[32];
    float pmax = -1e30f;
    const int qrow = q0 + r;
#pragma unroll
    for (int c = 0; c < 32; ++c) {
      int col = hc * 32 + c;
      float s = Ssc[w][r][col];
      if (kt * 64 + col > qrow) s = -1e30f;
      sv[c] = s;
      pmax = fmaxf(pmax, s);
    }
    pmax = fmaxf(pmax, __shfl_xor(pmax, 16, 32));
    const float m_new = fmaxf(m_run, pmax);
    const float corr = __expf(m_run - m_new);
    float psum = 0.f;
#pragma unroll
    for (int c = 0; c < 32; ++c) {
      float e = __expf(sv[c] - m_new);
      psum += e;
      sv[c] = e;
    }
    psum += __shfl_xor(psum, 16, 32);
    l_run = l_run * corr + psum;
    m_run = m_new;
    if (hc == 0) corrw[w][r] = corr;

    bf16* P = (bf16*)&Ssc[w][0][0];              // re-use as bf16 P [16][64]
#pragma unroll
    for (int i = 0; i < 4; ++i) {                // packed 16B stores
      v8bf pk;
#pragma unroll
      for (int jj = 0; jj < 8; ++jj) pk[jj] = (bf16)sv[i * 8 + jj];
      *reinterpret_cast<v8bf*>(&P[r * 64 + hc * 32 + i * 8]) = pk;
    }

    // ---- O = O*corr + P.V ----
#pragma unroll
    for (int dt = 0; dt < 4; ++dt)
#pragma unroll
      for (int e = 0; e < 8; ++e) accO[dt][e] *= corrw[w][e + 8 * hc];
    v16bf ap0 = load_a_frag(P, 0, 64, 0);
    v16bf ap1 = load_a_frag(P, 0, 64, 32);
    {
      v16bf bv[8];
#pragma unroll
      for (int dt = 0; dt < 4; ++dt) {
        bv[2 * dt]     = load_b_frag(&Vs[0][0], dt * 16, 64, 0);
        bv[2 * dt + 1] = load_b_frag(&Vs[0][0], dt * 16, 64, 32);
      }
#pragma unroll
      for (int dt = 0; dt < 4; ++dt) {
        accO[dt] = wmma_bf16(ap0, bv[2 * dt], accO[dt]);
        accO[dt] = wmma_bf16(ap1, bv[2 * dt + 1], accO[dt]);
      }
    }
  }

  if (hc == 0) lw[w][r] = l_run;
#pragma unroll
  for (int dt = 0; dt < 4; ++dt)
#pragma unroll
    for (int e = 0; e < 8; ++e) {
      int rowM = e + 8 * hc;
      int t = q0 + rowM;
      int d = dt * 16 + n16;
      float outv = accO[dt][e] / lw[w][rowM];
      size_t gidx = ((size_t)b * kT + t) * kD + hh * 64 + d;
      h[gidx] = x_ori[gidx] + outv;
    }
}

// ---------------------------------------------------------------------------
// FFN stage 1 fused: g = silu(hn@w1 + b1) * (hn@w3 + b3)  (bf16 out)
// Double-buffered async-LDS staging. grid = (Tok/128, F/128), block 256.
// ---------------------------------------------------------------------------
__global__ void __launch_bounds__(256) k_ffn1(const bf16* __restrict__ hn,
                                              const bf16* __restrict__ w1t,
                                              const bf16* __restrict__ w3t,
                                              const float* __restrict__ b1,
                                              const float* __restrict__ b3,
                                              bf16* __restrict__ g) {
  __shared__ bf16 As[2][128][32];
  __shared__ bf16 B1s[2][128][32];
  __shared__ bf16 B3s[2][128][32];
  const int tid = threadIdx.x, w = tid >> 5, lane = tid & 31;
  const int n16 = lane & 15, hc = lane >> 4;
  const int m0 = blockIdx.x * 128, n0 = blockIdx.y * 128;

  auto stage = [&](int buf, int k0) {
#pragma unroll
    for (int l = 0; l < 2; ++l) {
      int ch = tid + l * 256;
      int r = ch >> 2, c8 = (ch & 3) * 8;
      async_cp16(&As[buf][r][c8],  &hn[(size_t)(m0 + r) * kD + k0 + c8]);
      async_cp16(&B1s[buf][r][c8], &w1t[(size_t)(n0 + r) * kD + k0 + c8]);
      async_cp16(&B3s[buf][r][c8], &w3t[(size_t)(n0 + r) * kD + k0 + c8]);
    }
  };

  v8f a1[8], a3[8];
#pragma unroll
  for (int i = 0; i < 8; ++i) { a1[i] = zero_v8f(); a3[i] = zero_v8f(); }

  stage(0, 0);
  async_wait();
  __syncthreads();
  constexpr int nIt = kD / 32;
  for (int it = 0; it < nIt; ++it) {
    const int cur = it & 1;
    if (it + 1 < nIt) stage(1 - cur, (it + 1) * 32);
    v16bf a = load_a_frag(&As[cur][0][0], w * 16, 32, 0);
    v16bf bk[4];
#pragma unroll
    for (int nt = 0; nt < 4; ++nt) bk[nt] = load_b_frag(&B1s[cur][0][0], nt * 16, 32, 0);
#pragma unroll
    for (int nt = 0; nt < 4; ++nt) a1[nt] = wmma_bf16(a, bk[nt], a1[nt]);
#pragma unroll
    for (int nt = 0; nt < 4; ++nt) bk[nt] = load_b_frag(&B1s[cur][0][0], (nt + 4) * 16, 32, 0);
#pragma unroll
    for (int nt = 0; nt < 4; ++nt) a1[nt + 4] = wmma_bf16(a, bk[nt], a1[nt + 4]);
#pragma unroll
    for (int nt = 0; nt < 4; ++nt) bk[nt] = load_b_frag(&B3s[cur][0][0], nt * 16, 32, 0);
#pragma unroll
    for (int nt = 0; nt < 4; ++nt) a3[nt] = wmma_bf16(a, bk[nt], a3[nt]);
#pragma unroll
    for (int nt = 0; nt < 4; ++nt) bk[nt] = load_b_frag(&B3s[cur][0][0], (nt + 4) * 16, 32, 0);
#pragma unroll
    for (int nt = 0; nt < 4; ++nt) a3[nt + 4] = wmma_bf16(a, bk[nt], a3[nt + 4]);
    async_wait();
    __syncthreads();
  }
#pragma unroll
  for (int nt = 0; nt < 8; ++nt)
#pragma unroll
    for (int e = 0; e < 8; ++e) {
      int row = m0 + w * 16 + e + 8 * hc;
      int col = n0 + nt * 16 + n16;
      float x1 = a1[nt][e] + b1[col];
      float x3 = a3[nt][e] + b3[col];
      float si = x1 / (1.f + __expf(-x1));
      g[(size_t)row * kF + col] = (bf16)(si * x3);
    }
}

// ---------------------------------------------------------------------------
// FFN stage 2: out = g @ w2 + b2 + h  (fp32 out, residual fused)
// Double-buffered async-LDS staging. grid = (Tok/128, D/128), block 256.
// ---------------------------------------------------------------------------
__global__ void __launch_bounds__(256) k_ffn2(const bf16* __restrict__ g,
                                              const bf16* __restrict__ w2t,
                                              const float* __restrict__ b2,
                                              const float* __restrict__ h,
                                              float* __restrict__ out) {
  __shared__ bf16 As[2][128][32];
  __shared__ bf16 Bs[2][128][32];
  const int tid = threadIdx.x, w = tid >> 5, lane = tid & 31;
  const int n16 = lane & 15, hc = lane >> 4;
  const int m0 = blockIdx.x * 128, n0 = blockIdx.y * 128;

  auto stage = [&](int buf, int k0) {
#pragma unroll
    for (int l = 0; l < 2; ++l) {
      int ch = tid + l * 256;
      int r = ch >> 2, c8 = (ch & 3) * 8;
      async_cp16(&As[buf][r][c8], &g[(size_t)(m0 + r) * kF + k0 + c8]);
      async_cp16(&Bs[buf][r][c8], &w2t[(size_t)(n0 + r) * kF + k0 + c8]);
    }
  };

  v8f acc[8];
#pragma unroll
  for (int i = 0; i < 8; ++i) acc[i] = zero_v8f();

  stage(0, 0);
  async_wait();
  __syncthreads();
  constexpr int nIt = kF / 32;
  for (int it = 0; it < nIt; ++it) {
    const int cur = it & 1;
    if (it + 1 < nIt) stage(1 - cur, (it + 1) * 32);
    v16bf a = load_a_frag(&As[cur][0][0], w * 16, 32, 0);
    v16bf bk[4];
#pragma unroll
    for (int nt = 0; nt < 4; ++nt) bk[nt] = load_b_frag(&Bs[cur][0][0], nt * 16, 32, 0);
#pragma unroll
    for (int nt = 0; nt < 4; ++nt) acc[nt] = wmma_bf16(a, bk[nt], acc[nt]);
#pragma unroll
    for (int nt = 0; nt < 4; ++nt) bk[nt] = load_b_frag(&Bs[cur][0][0], (nt + 4) * 16, 32, 0);
#pragma unroll
    for (int nt = 0; nt < 4; ++nt) acc[nt + 4] = wmma_bf16(a, bk[nt], acc[nt + 4]);
    async_wait();
    __syncthreads();
  }
#pragma unroll
  for (int nt = 0; nt < 8; ++nt)
#pragma unroll
    for (int e = 0; e < 8; ++e) {
      int row = m0 + w * 16 + e + 8 * hc;
      int col = n0 + nt * 16 + n16;
      out[(size_t)row * kD + col] = acc[nt][e] + b2[col] + h[(size_t)row * kD + col];
    }
}

// ---------------------------------------------------------------------------
// Host launcher
// ---------------------------------------------------------------------------
extern "C" void kernel_launch(void* const* d_in, const int* in_sizes, int n_in,
                              void* d_out, int out_size, void* d_ws, size_t ws_size,
                              hipStream_t stream) {
  (void)in_sizes; (void)n_in; (void)out_size; (void)ws_size;
  const float* x_ori = (const float*)d_in[0];
  const float* wq = (const float*)d_in[1];
  const float* wk = (const float*)d_in[2];
  const float* wv = (const float*)d_in[3];
  const float* w1 = (const float*)d_in[4];
  const float* b1 = (const float*)d_in[5];
  const float* w2 = (const float*)d_in[6];
  const float* b2 = (const float*)d_in[7];
  const float* w3 = (const float*)d_in[8];
  const float* b3 = (const float*)d_in[9];
  float* out = (float*)d_out;

  char* p = (char*)d_ws;
  auto carve = [&](size_t bytes) -> void* {
    void* r = (void*)p;
    p += (bytes + 255) & ~(size_t)255;
    return r;
  };
  const size_t nAct = (size_t)kTok * kD;
  const size_t nQkvW = (size_t)kH * kD * kHd;
  const size_t nHeads = (size_t)kB * kH * kT * kHd;
  const size_t nFfnW = (size_t)kD * kF;
  const size_t nG = (size_t)kTok * kF;

  bf16* xn  = (bf16*)carve(nAct * 2);
  bf16* wqt = (bf16*)carve(nQkvW * 2);
  bf16* wkt = (bf16*)carve(nQkvW * 2);
  bf16* wvt = (bf16*)carve(nQkvW * 2);
  bf16* qb  = (bf16*)carve(nHeads * 2);
  bf16* kb  = (bf16*)carve(nHeads * 2);
  bf16* vtb = (bf16*)carve(nHeads * 2);
  float* hbuf = (float*)carve(nAct * 4);
  bf16* hn  = (bf16*)carve(nAct * 2);
  bf16* w1t = (bf16*)carve(nFfnW * 2);
  bf16* w3t = (bf16*)carve(nFfnW * 2);
  bf16* w2t = (bf16*)carve(nFfnW * 2);
  bf16* g   = (bf16*)carve(nG * 2);

  // weight conversion + transpose to [n][k]
  k_transpose_bf16<<<dim3(kD / 32, kHd / 32, kH), 256, 0, stream>>>(wq, wqt, kD, kHd);
  k_transpose_bf16<<<dim3(kD / 32, kHd / 32, kH), 256, 0, stream>>>(wk, wkt, kD, kHd);
  k_transpose_bf16<<<dim3(kD / 32, kHd / 32, kH), 256, 0, stream>>>(wv, wvt, kD, kHd);
  k_transpose_bf16<<<dim3(kD / 32, kF / 32, 1), 256, 0, stream>>>(w1, w1t, kD, kF);
  k_transpose_bf16<<<dim3(kD / 32, kF / 32, 1), 256, 0, stream>>>(w3, w3t, kD, kF);
  k_transpose_bf16<<<dim3(kF / 32, kD / 32, 1), 256, 0, stream>>>(w2, w2t, kF, kD);

  // attention_norm
  k_rmsnorm_bf16<<<kTok, 256, 0, stream>>>(x_ori, xn);
  // QKV projections (V stored transposed)
  k_qkv<<<dim3(kT / 128, 3, kB * kH), 256, 0, stream>>>(xn, wqt, wkt, wvt, qb, kb, vtb);
  // causal attention + residual -> h
  k_attn<<<dim3(kT / 128, 1, kB * kH), 256, 0, stream>>>(qb, kb, vtb, x_ori, hbuf);
  // ffn_norm
  k_rmsnorm_bf16<<<kTok, 256, 0, stream>>>(hbuf, hn);
  // SwiGLU up projections fused with activation
  k_ffn1<<<dim3(kTok / 128, kF / 128), 256, 0, stream>>>(hn, w1t, w3t, b1, b3, g);
  // down projection + bias + residual
  k_ffn2<<<dim3(kTok / 128, kD / 128), 256, 0, stream>>>(g, w2t, b2, hbuf, out);
}
